// OperationFeatureBlock_57939108823286
// MI455X (gfx1250) — matrix operand order repaired and checked by reading, not compiled, not run on hardware
//
#include <hip/hip_runtime.h>

// ---------------------------------------------------------------------------
// OperationFeatureBlock on MI455X (gfx1250, wave32, WMMA).
// All GEMMs run on V_WMMA_F32_16X16X32_F16 (f16 in, f32 accumulate).
// B-operands (weights + encoder message features) live in "fragment-major"
// layout so every fragment load is 2x ds_load_b128 / global_store_b128.
// Masks are JAX bools -> raw 1-byte 0/1 on device (uint8).
// ---------------------------------------------------------------------------

typedef __attribute__((ext_vector_type(16))) _Float16 v16h;
typedef __attribute__((ext_vector_type(8)))  float    v8f;

#define B_   8
#define C_   2048
#define F_   128
#define NR_  (B_ * C_)   // 16384 rows flattened
#define TK   32          // K tile depth (16x16x32 WMMA)
#define BAND 4096        // halves per K-band in frag-major (8 nt * 32 * 16)

// frag-major address of element (k, n) of a [K][128] matrix:
// band = k>>5, nt = n>>4, lane = (n&15) + 16*((k>>4)&1), half = k&15
__device__ __forceinline__ size_t frag_idx(int k, int n) {
  return ((size_t)((((k >> 5) * 8 + (n >> 4)) * 32) +
                   (n & 15) + (((k >> 4) & 1) << 4)) << 4) + (size_t)(k & 15);
}

// A fragment (16x32 f16) from row-major LDS tile (contiguous k = 16h + s, same order as frag-major B):
// lanes 0-15: M=lane, K in {0..15}; lanes 16-31: M=lane-16, K in {16..31}
__device__ __forceinline__ v16h a_frag(const _Float16* a, int stride, int lane) {
  const int r  = lane & 15;
  const int k0 = (lane >> 4) << 4;
  const _Float16* row = a + r * stride + k0;
  v16h f;
#pragma unroll
  for (int i = 0; i < 16; ++i) { f[i] = row[i]; }
  return f;
}

// --- f32 -> f16 row-major conversion ---------------------------------------
__global__ void f32_to_f16_kernel(const float* __restrict__ s,
                                  _Float16* __restrict__ d, int n) {
  int i = blockIdx.x * blockDim.x + threadIdx.x;
  if (i < n) d[i] = (_Float16)s[i];
}

// --- f32 weights -> f16 frag-major pack ------------------------------------
__global__ void pack_bfrag_kernel(const float* __restrict__ src, int sstride,
                                  int coff, _Float16* __restrict__ dst,
                                  int total /* K*128 */) {
  int i = blockIdx.x * blockDim.x + threadIdx.x;
  if (i >= total) return;
  int k = i >> 7, n = i & 127;
  dst[frag_idx(k, n)] = (_Float16)src[(size_t)k * sstride + coff + n];
}

// --- Generic WMMA GEMM: A = row-major concat of up to 3 [rows][128] sources,
//     B = frag-major [nkt bands].  Double-buffered LDS tiles. --------------
__global__ __launch_bounds__(256) void wmma_gemm_concat(
    const _Float16* __restrict__ s0, const _Float16* __restrict__ s1,
    const _Float16* __restrict__ s2, int nkt,
    const _Float16* __restrict__ Bfrag,
    const float* __restrict__ bias, int boff,
    _Float16* __restrict__ o16, int o16_frag, float* __restrict__ o32) {
  __shared__ _Float16 As[2][128][TK];   // 2 x 8 KB
  __shared__ _Float16 Bs[2][BAND];      // 2 x 8 KB
  const int tid = threadIdx.x, wid = tid >> 5, lane = tid & 31;
  const size_t row0 = (size_t)blockIdx.x * 128;
  const _Float16* srcs[3] = { s0, s1, s2 };

  auto stage = [&](int kt, int buf) {
    { // A tile: 128 rows x 32 halves, 2 threads/row, 32B each
      const _Float16* src = srcs[kt >> 2];
      const int koff = (kt & 3) * TK;
      const int r = tid >> 1, cb = (tid & 1) * 16;
      const uint4* g = (const uint4*)(src + (row0 + r) * F_ + koff + cb);
      uint4* l = (uint4*)&As[buf][r][cb];
      l[0] = g[0]; l[1] = g[1];
    }
    { // B band: contiguous 8 KB copy (frag-major upstream)
      const uint4* g = (const uint4*)(Bfrag + (size_t)kt * BAND);
      uint4* l = (uint4*)&Bs[buf][0];
      l[2 * tid] = g[2 * tid]; l[2 * tid + 1] = g[2 * tid + 1];
    }
  };

  v8f acc[8] = {};
  stage(0, 0);
  for (int kt = 0; kt < nkt; ++kt) {
    __syncthreads();
    if (kt + 1 < nkt) stage(kt + 1, (kt + 1) & 1);
    const int buf = kt & 1;
    v16h af = a_frag(&As[buf][wid * 16][0], TK, lane);
    const v16h* bp = (const v16h*)&Bs[buf][0];
#pragma unroll
    for (int nt = 0; nt < 8; ++nt) {
      v16h bf = bp[nt * 32 + lane];
      acc[nt] = __builtin_amdgcn_wmma_f32_16x16x32_f16(
          false, af, false, bf, (short)0, acc[nt], false, false);
    }
  }

  // epilogue: VGPR j holds M=j (lanes 0-15) / M=j+8 (lanes 16-31)
  const int n0 = lane & 15, mhi = (lane >> 4) * 8;
  if (o32) {
#pragma unroll
    for (int nt = 0; nt < 8; ++nt) {
      const int col = nt * 16 + n0; const float bv = bias[boff + col];
#pragma unroll
      for (int j = 0; j < 8; ++j)
        o32[(row0 + wid * 16 + mhi + j) * F_ + col] = acc[nt][j] + bv;
    }
  } else if (o16_frag) {
    // fragment-major store: 8 j-values are contiguous halves -> one b128/nt
    const size_t ktOut = (row0 >> 5) + (wid >> 1);
#pragma unroll
    for (int nt = 0; nt < 8; ++nt) {
      const int col = nt * 16 + n0; const float bv = bias[boff + col];
      union { _Float16 h[8]; uint4 u; } pk;
#pragma unroll
      for (int j = 0; j < 8; ++j) pk.h[j] = (_Float16)(acc[nt][j] + bv);
      const size_t haddr =
          (((ktOut * 8 + nt) * 32) + n0 + ((wid & 1) << 4)) * 16 + mhi;
      *(uint4*)(o16 + haddr) = pk.u;
    }
  } else {
#pragma unroll
    for (int nt = 0; nt < 8; ++nt) {
      const int col = nt * 16 + n0; const float bv = bias[boff + col];
#pragma unroll
      for (int j = 0; j < 8; ++j)
        o16[(row0 + wid * 16 + mhi + j) * F_ + col] =
            (_Float16)(acc[nt][j] + bv);
    }
  }
}

// --- Masked mean: out[b] = (mask[b] @ e[b]) / max(rowsum(mask[b]),1) -------
// e is frag-major; row degrees folded into mask->f16 conversion via ds_add.
__global__ __launch_bounds__(256) void wmma_masked_mean(
    const unsigned char* __restrict__ mask,   // [B][C][C] 0/1 bytes
    const _Float16* __restrict__ efrag,       // [B][C][F] frag-major
    _Float16* __restrict__ o) {               // [B][C][F] row-major
  __shared__ _Float16 As[2][128][TK];
  __shared__ _Float16 Bs[2][BAND];
  __shared__ int deg[128];
  const int tid = threadIdx.x, wid = tid >> 5, lane = tid & 31;
  const size_t mbase = (size_t)blockIdx.y * C_ * C_;
  const size_t ebase = (size_t)blockIdx.y * C_ * F_;
  const size_t row0  = (size_t)blockIdx.x * 128;

  auto stage = [&](int kt, int buf) {
    { // mask bytes -> 16 halves in regs -> 2x ds_store_b128; fold degree
      const int r = tid >> 1, cb = (tid & 1) * 16;
      const uint4 m =
          *(const uint4*)(mask + mbase + (row0 + r) * C_ + kt * TK + cb);
      const unsigned char* mb = (const unsigned char*)&m;
      union { _Float16 h[16]; uint4 q[2]; } cv;
      int s = 0;
#pragma unroll
      for (int i = 0; i < 16; ++i) { s += mb[i]; cv.h[i] = (_Float16)(float)mb[i]; }
      uint4* l = (uint4*)&As[buf][r][cb];
      l[0] = cv.q[0]; l[1] = cv.q[1];
      if (s) atomicAdd(&deg[r], s);
    }
    { // e band: contiguous 8 KB copy
      const uint4* g = (const uint4*)(efrag + ebase + (size_t)kt * BAND);
      uint4* l = (uint4*)&Bs[buf][0];
      l[2 * tid] = g[2 * tid]; l[2 * tid + 1] = g[2 * tid + 1];
    }
  };

  if (tid < 128) deg[tid] = 0;
  __syncthreads();

  v8f acc[8] = {};
  stage(0, 0);
  for (int kt = 0; kt < C_ / TK; ++kt) {
    __syncthreads();
    if (kt + 1 < C_ / TK) stage(kt + 1, (kt + 1) & 1);
    const int buf = kt & 1;
    v16h af = a_frag(&As[buf][wid * 16][0], TK, lane);
    const v16h* bp = (const v16h*)&Bs[buf][0];
#pragma unroll
    for (int nt = 0; nt < 8; ++nt) {
      v16h bf = bp[nt * 32 + lane];
      acc[nt] = __builtin_amdgcn_wmma_f32_16x16x32_f16(
          false, af, false, bf, (short)0, acc[nt], false, false);
    }
  }

  const int n0 = lane & 15, mhi = (lane >> 4) * 8;
#pragma unroll
  for (int j = 0; j < 8; ++j) {
    const int rl = wid * 16 + mhi + j;
    const int d  = deg[rl];
    const float inv = 1.0f / (float)(d > 0 ? d : 1);
#pragma unroll
    for (int nt = 0; nt < 8; ++nt)
      o[ebase + (row0 + rl) * F_ + nt * 16 + n0] = (_Float16)(acc[nt][j] * inv);
  }
}

// ---------------------------------------------------------------------------
extern "C" void kernel_launch(void* const* d_in, const int* in_sizes, int n_in,
                              void* d_out, int out_size, void* d_ws,
                              size_t ws_size, hipStream_t stream) {
  (void)in_sizes; (void)n_in; (void)out_size; (void)ws_size;
  const float*         features  = (const float*)d_in[0];
  const unsigned char* pred_mask = (const unsigned char*)d_in[1];
  const unsigned char* succ_mask = (const unsigned char*)d_in[2];
  const unsigned char* same_mask = (const unsigned char*)d_in[3];
  const float* W_enc   = (const float*)d_in[4];
  const float* b_enc   = (const float*)d_in[5];
  const float* W_space = (const float*)d_in[6];
  const float* b_space = (const float*)d_in[7];
  const float* W_same  = (const float*)d_in[8];
  const float* b_same  = (const float*)d_in[9];
  const float* W_mix   = (const float*)d_in[10];
  const float* b_mix   = (const float*)d_in[11];

  // workspace carve-up (f16 elements), ~42 MB total
  const size_t NRF = (size_t)NR_ * F_;          // 2,097,152
  _Float16* h        = (_Float16*)d_ws;
  _Float16* feat16   = h;                       // row-major  (A source)
  _Float16* origin16 = feat16   + NRF;          // row-major  (A source)
  _Float16* epred16  = origin16 + NRF;          // frag-major (B source)
  _Float16* esucc16  = epred16  + NRF;          // frag-major
  _Float16* esame16  = esucc16  + NRF;          // frag-major
  _Float16* predf16  = esame16  + NRF;          // row-major  (A source)
  _Float16* succf16  = predf16  + NRF;          // row-major
  _Float16* samef16  = succf16  + NRF;          // row-major
  _Float16* space16  = samef16  + NRF;          // row-major  (A source)
  _Float16* samem16  = space16  + NRF;          // row-major
  _Float16* wenc16   = samem16  + NRF;          // 4 x [128][128] frag-major
  _Float16* wspace16 = wenc16   + 128 * 512;    // [384][128]  frag-major
  _Float16* wsame16  = wspace16 + 384 * 128;    // [256][128]  frag-major
  _Float16* wmix16   = wsame16  + 256 * 128;    // [256][128]  frag-major

  // 1) conversions / weight packing (tiny vs. 100 MB of mask traffic)
  f32_to_f16_kernel<<<(NR_ * F_ + 255) / 256, 256, 0, stream>>>(
      features, feat16, NR_ * F_);
  for (int cb = 0; cb < 4; ++cb)
    pack_bfrag_kernel<<<(128 * 128 + 255) / 256, 256, 0, stream>>>(
        W_enc, 512, cb * 128, wenc16 + (size_t)cb * 128 * 128, 128 * 128);
  pack_bfrag_kernel<<<(384 * 128 + 255) / 256, 256, 0, stream>>>(
      W_space, 128, 0, wspace16, 384 * 128);
  pack_bfrag_kernel<<<(256 * 128 + 255) / 256, 256, 0, stream>>>(
      W_same, 128, 0, wsame16, 256 * 128);
  pack_bfrag_kernel<<<(256 * 128 + 255) / 256, 256, 0, stream>>>(
      W_mix, 128, 0, wmix16, 256 * 128);

  const dim3 g(NR_ / 128);   // 128 workgroups of 256 threads (8 waves)

  // 2) encoder: origin row-major; e_pred/e_succ/e_same frag-major (B sources)
  _Float16* encOut[4] = { origin16, epred16, esucc16, esame16 };
  for (int cb = 0; cb < 4; ++cb)
    wmma_gemm_concat<<<g, 256, 0, stream>>>(
        feat16, nullptr, nullptr, /*nkt=*/4,
        wenc16 + (size_t)cb * 128 * 128,
        b_enc, cb * 128, encOut[cb], /*frag=*/cb != 0, nullptr);

  // 3) three masked means (mask @ e) / clamp(deg,1)
  const dim3 gm(C_ / 128, B_);
  wmma_masked_mean<<<gm, 256, 0, stream>>>(pred_mask, epred16, predf16);
  wmma_masked_mean<<<gm, 256, 0, stream>>>(succ_mask, esucc16, succf16);
  wmma_masked_mean<<<gm, 256, 0, stream>>>(same_mask, esame16, samef16);

  // 4) space_mixed = [pred_f, origin, succ_f] @ W_space + b   (K = 384)
  wmma_gemm_concat<<<g, 256, 0, stream>>>(
      predf16, origin16, succf16, 12, wspace16, b_space, 0,
      space16, 0, nullptr);

  // 5) same_mixed = [origin, same_f] @ W_same + b             (K = 256)
  wmma_gemm_concat<<<g, 256, 0, stream>>>(
      origin16, samef16, nullptr, 8, wsame16, b_same, 0,
      samem16, 0, nullptr);

  // 6) out = [space_mixed, same_mixed] @ W_mix + b  (K = 256, f32 out)
  wmma_gemm_concat<<<g, 256, 0, stream>>>(
      space16, samem16, nullptr, 8, wmix16, b_mix, 0,
      nullptr, 0, (float*)d_out);
}
